// GraphTransformer_3813930959277
// MI455X (gfx1250) — matrix operand order repaired and checked by reading, not compiled, not run on hardware
//
#include <hip/hip_runtime.h>

// ---------------------------------------------------------------------------
// Graph Transformer for MI455X (gfx1250).
// Dense GEMMs -> v_wmma_f32_16x16x32_bf16 (f32 accumulate); B staged into LDS
// with global_load_async_to_lds_b128 (ASYNCcnt path, no VGPR bounce).
// Sparse attention phase -> fp32 gathers + global f32 atomics (memory bound,
// ~1GB/layer random traffic @ 23.3 TB/s dominates; the 43 GFLOP of dense work
// hides behind it once it runs on the matrix cores).
// ---------------------------------------------------------------------------

#define NN   50000
#define DD   128
#define HH   8
#define EE   500000
#define FFN  512
#define LL   2

typedef __bf16        v16bf __attribute__((ext_vector_type(16)));
typedef float         v8f   __attribute__((ext_vector_type(8)));
typedef unsigned int  u32x4 __attribute__((ext_vector_type(4)));

union FragB { v16bf v; u32x4 q[2]; };

static __device__ __forceinline__ unsigned short f2bf(float f) {
  unsigned int u = __float_as_uint(f);
  unsigned int r = u + 0x7fffu + ((u >> 16) & 1u);
  return (unsigned short)(r >> 16);
}

// ---------------------------------------------------------------------------
// Epilogue modes
// ---------------------------------------------------------------------------
#define EPI_F32   0   // outF = acc + bias
#define EPI_XP    1   // outB = bf16(resid + acc + bias + degs[row]*wdeg[col] + bdeg[col])
#define EPI_RESID 2   // outF = resid + acc + bias
#define EPI_GELU  3   // outB = bf16(gelu(acc + bias))

// ---------------------------------------------------------------------------
// Tiled WMMA GEMM:  C[M x (gridDim.y*128)] = A[M x K](bf16) @ Bt^T
//   A   : bf16 row-major [M x K]
//   Bt  : bf16 row-major [Ncols x K]  (i.e. B transposed)
//   K multiple of 128.  block = 256 threads = 8 waves; wave -> 16 rows x 128 cols.
// ---------------------------------------------------------------------------
__global__ void __launch_bounds__(256)
gemm_bf16(const unsigned short* __restrict__ A,
          const unsigned short* __restrict__ Bt,
          int M, int K,
          const float* __restrict__ bias,
          const float* __restrict__ resid,
          const float* __restrict__ wdeg,
          const float* __restrict__ bdeg,
          const float* __restrict__ degs,
          float* __restrict__ outF,
          unsigned short* __restrict__ outB,
          int mode)
{
  __shared__ unsigned short ldsB[128 * 136];   // Bt chunk [128 cols][128 k] (+pad)

  const int wave = threadIdx.x >> 5;
  const int lane = threadIdx.x & 31;
  const int hf   = lane >> 4;        // lane half
  const int l15  = lane & 15;
  const int tile = blockIdx.x * 8 + wave;
  const int rowBase = tile * 16;
  const int colChunk = blockIdx.y;
  const int strideN = gridDim.y * 128;

  v8f acc[8];
  v8f z = {0.f,0.f,0.f,0.f,0.f,0.f,0.f,0.f};
#pragma unroll
  for (int i = 0; i < 8; ++i) acc[i] = z;

  const unsigned long long btBase = (unsigned long long)Bt;

  for (int kc = 0; kc < K; kc += 128) {
    __syncthreads();                 // prior-chunk readers done before overwrite
    // Async direct-to-LDS staging of Bt chunk: 128 cols x 128 k (bf16).
    // GVS mode: SGPR base + per-lane byte offset; VDST = LDS byte offset.
#pragma unroll
    for (int i = 0; i < 8; ++i) {
      int idx = threadIdx.x + i * 256;            // 0..2047
      int n   = idx >> 4;                          // 0..127
      int kk  = (idx & 15) << 3;                   // 0..120 step 8
      unsigned ldsOff = (unsigned)(unsigned long long)&ldsB[n * 136 + kk];
      unsigned gOff   = (unsigned)((((unsigned)(colChunk * 128 + n)) * (unsigned)K
                                    + (unsigned)(kc + kk)) * 2u);
      asm volatile("global_load_async_to_lds_b128 %0, %1, %2"
                   :: "v"(ldsOff), "v"(gOff), "s"(btBase)
                   : "memory");
    }
    asm volatile("s_wait_asynccnt 0x0" ::: "memory");
    __syncthreads();

    int arow = rowBase + l15; if (arow >= M) arow = M - 1;   // clamp (store-guarded)
    const unsigned short* Abase = A + (size_t)arow * K + kc + hf * 8;

#pragma unroll
    for (int kt = 0; kt < 4; ++kt) {
      // A frag (ISA 16-bit A 16x32): lane<16 -> K {0..7,16..23}; lane>=16 -> +8
      FragB af;
      af.q[0] = *(const u32x4*)(Abase + kt * 32);
      af.q[1] = *(const u32x4*)(Abase + kt * 32 + 16);
      // Load all 8 B frags first (one DS clause), then 8 WMMAs back-to-back.
      FragB bfr[8];
#pragma unroll
      for (int nt = 0; nt < 8; ++nt) {
        // B frag: lane<16 -> N=lane, K {0..7,16..23} ; lane>=16 -> N=lane-16, K {8..15,24..31}
        const unsigned short* Bb = &ldsB[(nt * 16 + l15) * 136 + kt * 32 + hf * 8];
        bfr[nt].q[0] = *(const u32x4*)(Bb);
        bfr[nt].q[1] = *(const u32x4*)(Bb + 16);
      }
#pragma unroll
      for (int nt = 0; nt < 8; ++nt) {
        acc[nt] = __builtin_amdgcn_wmma_f32_16x16x32_bf16(
            false, af.v, false, bfr[nt].v, (short)0, acc[nt], false, false);
      }
    }
  }

  // Epilogue (after all WMMA; per-lane guards only here)
  const int colBase = colChunk * 128;
#pragma unroll
  for (int nt = 0; nt < 8; ++nt) {
    int col = colBase + nt * 16 + l15;
    float bcol = bias[col];
#pragma unroll
    for (int j = 0; j < 8; ++j) {
      int row = rowBase + hf * 8 + j;
      if (row < M) {
        float val = acc[nt][j] + bcol;
        size_t off = (size_t)row * strideN + col;
        if (mode == EPI_XP) {
          val += resid[off] + degs[row] * wdeg[col] + bdeg[col];
          outB[off] = f2bf(val);
        } else if (mode == EPI_F32) {
          outF[off] = val;
        } else if (mode == EPI_RESID) {
          outF[off] = resid[off] + val;
        } else { // EPI_GELU (exact erf gelu)
          float g = 0.5f * val * (1.0f + erff(val * 0.70710678118654752f));
          outB[off] = f2bf(g);
        }
      }
    }
  }
}

// ---------------------------------------------------------------------------
// Weight transpose + fp32 -> bf16 :  W[K x Nc] -> Wt[Nc x K]
// ---------------------------------------------------------------------------
__global__ void wtrans_k(const float* __restrict__ W, unsigned short* __restrict__ Wt,
                         int K, int Nc)
{
  int t = blockIdx.x * 256 + threadIdx.x;
  if (t >= K * Nc) return;
  int k = t / Nc, n = t - k * Nc;
  Wt[(size_t)n * K + k] = f2bf(W[t]);
}

// ---------------------------------------------------------------------------
// LayerNorm over D=128, one block per row (128 threads)
// ---------------------------------------------------------------------------
__global__ void layernorm_k(const float* __restrict__ x,
                            const float* __restrict__ g,
                            const float* __restrict__ b,
                            float* __restrict__ outF,
                            unsigned short* __restrict__ outB, int N)
{
  __shared__ float red[128];
  int row = blockIdx.x, tid = threadIdx.x;
  float v = x[(size_t)row * DD + tid];
  red[tid] = v; __syncthreads();
  for (int s = 64; s > 0; s >>= 1) { if (tid < s) red[tid] += red[tid + s]; __syncthreads(); }
  float mean = red[0] / (float)DD; __syncthreads();
  float d = v - mean;
  red[tid] = d * d; __syncthreads();
  for (int s = 64; s > 0; s >>= 1) { if (tid < s) red[tid] += red[tid + s]; __syncthreads(); }
  float var = red[0] / (float)DD;
  float y = d * rsqrtf(var + 1e-5f) * g[tid] + b[tid];
  size_t off = (size_t)row * DD + tid;
  if (outF) outF[off] = y;
  if (outB) outB[off] = f2bf(y);
}

// ---------------------------------------------------------------------------
// misc elementwise kernels
// ---------------------------------------------------------------------------
__global__ void fill_f32(float* p, float v, int n) {
  int t = blockIdx.x * 256 + threadIdx.x;
  if (t < n) p[t] = v;
}

__global__ void cvt_bf16(const float* __restrict__ in, unsigned short* __restrict__ out, int n) {
  int t = blockIdx.x * 256 + threadIdx.x;
  if (t < n) out[t] = f2bf(in[t]);
}

// degree: deg initialized to 1.0 (self loop), count src occurrences
__global__ void deg_count(const int* __restrict__ src, float* __restrict__ deg, int E) {
  int t = blockIdx.x * 256 + threadIdx.x;
  if (t < E) atomicAdd(&deg[src[t]], 1.0f);
}

__global__ void deg_stats_k(const float* __restrict__ deg, float* __restrict__ stats, int N) {
  __shared__ float s1[256], s2[256];
  float a = 0.f, b = 0.f;
  for (int i = threadIdx.x; i < N; i += 256) { float d = deg[i]; a += d; b += d * d; }
  s1[threadIdx.x] = a; s2[threadIdx.x] = b; __syncthreads();
  for (int s = 128; s > 0; s >>= 1) {
    if (threadIdx.x < s) { s1[threadIdx.x] += s1[threadIdx.x + s]; s2[threadIdx.x] += s2[threadIdx.x + s]; }
    __syncthreads();
  }
  if (threadIdx.x == 0) {
    float mean = s1[0] / (float)N;
    float var  = (s2[0] - s1[0] * s1[0] / (float)N) / (float)(N - 1);   // ddof=1
    float sd   = sqrtf(fmaxf(var, 0.f));
    stats[0] = mean;
    stats[1] = 1.0f / (sd + 1e-6f);
  }
}

__global__ void deg_standardize(const float* __restrict__ deg, const float* __restrict__ stats,
                                float* __restrict__ degs, int N) {
  int t = blockIdx.x * 256 + threadIdx.x;
  if (t < N) degs[t] = (deg[t] - stats[0]) * stats[1];
}

// ---------------------------------------------------------------------------
// attention pass 1: per (edge,head) logit -> exp -> denominator atomics
// (scale-invariant softmax: att clipped to [-50,50] so exp() is fp32-safe;
//  identical result to max-subtracted segment softmax)
// ---------------------------------------------------------------------------
__global__ void attn_logits(const float* __restrict__ q, const float* __restrict__ k,
                            const int* __restrict__ src, const int* __restrict__ dst,
                            float* __restrict__ ex, float* __restrict__ sden, int E)
{
  int t = blockIdx.x * 256 + threadIdx.x;
  int total = (E + NN) * HH;
  if (t >= total) return;
  int e = t >> 3, h = t & 7;
  int si = (e < E) ? src[e] : (e - E);
  int di = (e < E) ? dst[e] : (e - E);
  const float* qp = q + (size_t)di * DD + h * 16;
  const float* kp = k + (size_t)si * DD + h * 16;
  float a = 0.f;
#pragma unroll
  for (int j = 0; j < 16; ++j) a += qp[j] * kp[j];
  a *= 0.25f;                                  // / sqrt(16)
  if (!(a == a)) a = 0.0f;                     // nan -> 0
  a = fminf(fmaxf(a, -50.0f), 50.0f);          // clip (handles +/-inf too)
  float v = expf(a);
  ex[t] = v;
  atomicAdd(&sden[(size_t)di * HH + h], v);
}

// pass 2: alpha-weighted aggregation of v into destination rows
__global__ void attn_aggr(const float* __restrict__ vsrc,
                          const int* __restrict__ src, const int* __restrict__ dst,
                          const float* __restrict__ ex, const float* __restrict__ sden,
                          float* __restrict__ aggr, int E)
{
  int t = blockIdx.x * 256 + threadIdx.x;
  int total = (E + NN) * HH;
  if (t >= total) return;
  int e = t >> 3, h = t & 7;
  int si = (e < E) ? src[e] : (e - E);
  int di = (e < E) ? dst[e] : (e - E);
  float alpha = ex[t] / (sden[(size_t)di * HH + h] + 1e-16f);
  const float* vp = vsrc + (size_t)si * DD + h * 16;
  float* ap = aggr + (size_t)di * DD + h * 16;
#pragma unroll
  for (int j = 0; j < 16; ++j) atomicAdd(&ap[j], vp[j] * alpha);
}

// ---------------------------------------------------------------------------
// host-side orchestration
// ---------------------------------------------------------------------------
extern "C" void kernel_launch(void* const* d_in, const int* in_sizes, int n_in,
                              void* d_out, int out_size, void* d_ws, size_t ws_size,
                              hipStream_t stream)
{
  const float* x0   = (const float*)d_in[0];
  const int*   ei   = (const int*)  d_in[1];
  const float* Wq   = (const float*)d_in[2];
  const float* Wk   = (const float*)d_in[3];
  const float* Wv   = (const float*)d_in[4];
  const float* Wo   = (const float*)d_in[5];
  const float* Wpos = (const float*)d_in[6];
  const float* Wdeg = (const float*)d_in[7];
  const float* W1   = (const float*)d_in[8];
  const float* W2   = (const float*)d_in[9];
  const float* bq   = (const float*)d_in[10];
  const float* bk   = (const float*)d_in[11];
  const float* bv   = (const float*)d_in[12];
  const float* bo   = (const float*)d_in[13];
  const float* bpos = (const float*)d_in[14];
  const float* bdeg = (const float*)d_in[15];
  const float* ln1b = (const float*)d_in[16];
  const float* ln2b = (const float*)d_in[17];
  const float* b1   = (const float*)d_in[18];
  const float* b2   = (const float*)d_in[19];
  const float* ln1g = (const float*)d_in[20];
  const float* ln2g = (const float*)d_in[21];

  const int Etot = EE + NN;
  char* ws = (char*)d_ws;
  size_t off = 0;
  auto take = [&](size_t bytes) { size_t r = off; off += (bytes + 255) & ~(size_t)255; return r; };

  float*          xn   = (float*)(ws + take((size_t)NN * DD * 4));   // LN1 out; reused as aggr
  unsigned short* a0   = (unsigned short*)(ws + take((size_t)NN * DD * 2)); // bf16 A stage
  unsigned short* a1   = (unsigned short*)(ws + take((size_t)NN * DD * 2)); // xp bf16
  float*          qb   = (float*)(ws + take((size_t)NN * DD * 4));
  float*          kb   = (float*)(ws + take((size_t)NN * DD * 4));
  float*          vb   = (float*)(ws + take((size_t)NN * DD * 4));
  float*          exb  = (float*)(ws + take((size_t)Etot * HH * 4));
  float*          sden = (float*)(ws + take((size_t)NN * HH * 4));
  unsigned short* wt   = (unsigned short*)(ws + take((size_t)FFN * DD * 2));
  float*          deg  = (float*)(ws + take((size_t)NN * 4));
  float*          degs = (float*)(ws + take((size_t)NN * 4));
  float*          stats= (float*)(ws + take(256));
  unsigned short* hb   = (unsigned short*)qb;   // FFN hidden (bf16) aliases q+k (lifetimes disjoint)

  float* x = (float*)d_out;                     // residual stream lives in d_out

  hipMemcpyAsync(x, x0, (size_t)NN * DD * 4, hipMemcpyDeviceToDevice, stream);

  // ---- degree (layer-invariant): deg = 1 + count(src) ; standardize ddof=1
  fill_f32<<<(NN + 255) / 256, 256, 0, stream>>>(deg, 1.0f, NN);
  deg_count<<<(EE + 255) / 256, 256, 0, stream>>>(ei, deg, EE);
  deg_stats_k<<<1, 256, 0, stream>>>(deg, stats, NN);
  deg_standardize<<<(NN + 255) / 256, 256, 0, stream>>>(deg, stats, degs, NN);

  const int mTiles = (NN + 15) / 16;            // 3125
  const int gX = (mTiles + 7) / 8;              // 391
  dim3 gBlk(256);
  const int eGrid = (Etot * HH + 255) / 256;

  for (int l = 0; l < LL; ++l) {
    const float* Wq_l = Wq + (size_t)l * DD * DD;
    const float* Wk_l = Wk + (size_t)l * DD * DD;
    const float* Wv_l = Wv + (size_t)l * DD * DD;
    const float* Wo_l = Wo + (size_t)l * DD * DD;
    const float* Wp_l = Wpos + (size_t)l * DD * DD;
    const float* Wd_l = Wdeg + (size_t)l * DD;
    const float* W1_l = W1 + (size_t)l * DD * FFN;
    const float* W2_l = W2 + (size_t)l * FFN * DD;

    // LN1 -> xn (f32) + a0 (bf16)
    layernorm_k<<<NN, 128, 0, stream>>>(x, ln1g + l * DD, ln1b + l * DD, xn, a0, NN);

    // xp = xn + xn@Wpos + bpos + deg_std*Wdeg + bdeg   (bf16 out -> a1)
    wtrans_k<<<(DD * DD + 255) / 256, 256, 0, stream>>>(Wp_l, wt, DD, DD);
    gemm_bf16<<<dim3(gX, 1), gBlk, 0, stream>>>(a0, wt, NN, DD, bpos + l * DD,
        xn, Wd_l, bdeg + l * DD, degs, nullptr, a1, EPI_XP);

    // q/k/v = xp @ W* + b*   (f32 out)
    wtrans_k<<<(DD * DD + 255) / 256, 256, 0, stream>>>(Wq_l, wt, DD, DD);
    gemm_bf16<<<dim3(gX, 1), gBlk, 0, stream>>>(a1, wt, NN, DD, bq + l * DD,
        nullptr, nullptr, nullptr, nullptr, qb, nullptr, EPI_F32);
    wtrans_k<<<(DD * DD + 255) / 256, 256, 0, stream>>>(Wk_l, wt, DD, DD);
    gemm_bf16<<<dim3(gX, 1), gBlk, 0, stream>>>(a1, wt, NN, DD, bk + l * DD,
        nullptr, nullptr, nullptr, nullptr, kb, nullptr, EPI_F32);
    wtrans_k<<<(DD * DD + 255) / 256, 256, 0, stream>>>(Wv_l, wt, DD, DD);
    gemm_bf16<<<dim3(gX, 1), gBlk, 0, stream>>>(a1, wt, NN, DD, bv + l * DD,
        nullptr, nullptr, nullptr, nullptr, vb, nullptr, EPI_F32);

    // segment softmax + aggregation (aggr reuses xn buffer)
    fill_f32<<<(NN * HH + 255) / 256, 256, 0, stream>>>(sden, 0.0f, NN * HH);
    fill_f32<<<(NN * DD + 255) / 256, 256, 0, stream>>>(xn, 0.0f, NN * DD);
    attn_logits<<<eGrid, 256, 0, stream>>>(qb, kb, ei, ei + EE, exb, sden, EE);
    attn_aggr  <<<eGrid, 256, 0, stream>>>(vb, ei, ei + EE, exb, sden, xn, EE);

    // x += aggr @ Wo + bo
    cvt_bf16<<<(NN * DD + 255) / 256, 256, 0, stream>>>(xn, a0, NN * DD);
    wtrans_k<<<(DD * DD + 255) / 256, 256, 0, stream>>>(Wo_l, wt, DD, DD);
    gemm_bf16<<<dim3(gX, 1), gBlk, 0, stream>>>(a0, wt, NN, DD, bo + l * DD,
        x, nullptr, nullptr, nullptr, x, nullptr, EPI_RESID);

    // FFN: h = gelu(LN2(x) @ W1 + b1) ; x += h @ W2 + b2
    layernorm_k<<<NN, 128, 0, stream>>>(x, ln2g + l * DD, ln2b + l * DD, nullptr, a0, NN);
    wtrans_k<<<(DD * FFN + 255) / 256, 256, 0, stream>>>(W1_l, wt, DD, FFN);
    gemm_bf16<<<dim3(gX, 4), gBlk, 0, stream>>>(a0, wt, NN, DD, b1 + l * FFN,
        nullptr, nullptr, nullptr, nullptr, nullptr, hb, EPI_GELU);
    wtrans_k<<<(FFN * DD + 255) / 256, 256, 0, stream>>>(W2_l, wt, FFN, DD);
    gemm_bf16<<<dim3(gX, 1), gBlk, 0, stream>>>(hb, wt, NN, FFN, b2 + l * DD,
        x, nullptr, nullptr, nullptr, x, nullptr, EPI_RESID);
  }
}
